// SimpleGCNESAModel_20246475833771
// MI455X (gfx1250) — matrix operand order repaired and checked by reading, not compiled, not run on hardware
//
#include <hip/hip_runtime.h>

// ---------------------------------------------------------------------------
// CDNA5 (gfx1250): bf16 WMMA pipeline with pre-packed bf16 operands.
// Weights are staged into LDS per block with global_load_async_to_lds_b128.
// ---------------------------------------------------------------------------

typedef __attribute__((ext_vector_type(16))) __bf16 v16bf;
typedef __attribute__((ext_vector_type(8)))  __bf16 v8bf;
typedef __attribute__((ext_vector_type(8)))  float  v8f;

constexpr int   kH     = 128;
constexpr int   kNH    = 8;
constexpr int   kDH    = 16;
constexpr int   kBatch = 8;
constexpr int   kL     = 128;
constexpr int   kABlk  = 32;
constexpr int   kNBlk  = kBatch * kL;    // 1024
constexpr int   kNAt   = kNBlk * kABlk;  // 32768
constexpr int   kKA    = 10;
constexpr int   kKB    = 8;
constexpr int   kEB    = kL * kKB;       // 1024
constexpr int   kEdges = kNAt * kKA;     // 327680
constexpr float kEps   = 1e-5f;

__device__ __forceinline__ __bf16 cvt_bf16(float f) { return (__bf16)f; }

// Wave-local LDS fence: per-wave LDS round trips only need this wave's DS
// counter drained (wave32 lockstep).  Much cheaper than a block barrier.
__device__ __forceinline__ void wave_lds_fence() {
  asm volatile("s_wait_dscnt 0x0" ::: "memory");
}

// Async global->LDS copy of 16 bytes (ASYNCcnt tracked, no VGPR round trip).
__device__ __forceinline__ void async_copy_b128(unsigned lds_off,
                                                const void* gaddr) {
  asm volatile("global_load_async_to_lds_b128 %0, %1, off"
               :: "v"(lds_off), "v"((unsigned long long)(uintptr_t)gaddr)
               : "memory");
}
__device__ __forceinline__ void wait_asynccnt0() {
  asm volatile("s_wait_asynccnt 0x0" ::: "memory");
}

// ---------------------------------------------------------------------------
// GEMM: C[M,128] = alpha * (A[M,K](bf16) @ W[K,128] + bias), optional relu,
// output either f32 or bf16.  W pre-packed bf16 col-major: Wt[col*K + k].
// 8 waves/block, one 16x128 C strip per wave.
// K==128 path: the 32KB weight tile is staged into LDS once per block via
// async loads; all 8 waves then feed B fragments from LDS (ds_load_b128),
// removing the 8x redundant per-wave global weight traffic.
// ---------------------------------------------------------------------------
__global__ void __launch_bounds__(256)
gemm_wmma(const __bf16* __restrict__ A, int lda,
          const __bf16* __restrict__ Wt,
          const float* __restrict__ bias,
          void* __restrict__ Cv, int ldc,
          int M, int K, int relu, int store_bf16, float alpha)
{
  __shared__ __bf16 smw[16384];   // 32 KB weight stage (K==128 path)
  const int wave = threadIdx.x >> 5;
  const int lane = threadIdx.x & 31;
  const int h    = lane >> 4;
  const int r    = lane & 15;
  const int rowBase = (blockIdx.x * 8 + wave) * 16;

  const bool useLds = (K == 128);
  if (useLds) {
    // 256 threads x 16B x 8 rounds = 32KB
    unsigned lds0 = (unsigned)(uintptr_t)(&smw[0]) + threadIdx.x * 16u;
    const __bf16* g0 = Wt + threadIdx.x * 8;
#pragma unroll
    for (int it = 0; it < 8; ++it)
      async_copy_b128(lds0 + it * 4096u, g0 + it * 2048);
    wait_asynccnt0();
    __syncthreads();
  }
  if (rowBase >= M) return;   // never taken for our shapes (M % 128 == 0)

  v8f acc[8];
  v8f zero = {};
#pragma unroll
  for (int nt = 0; nt < 8; ++nt) acc[nt] = zero;

  for (int kb = 0; kb < K; kb += 32) {
    // A fragment: row r, K runs {kb+16h .. +15}
    const __bf16* arow = A + (size_t)(rowBase + r) * lda + kb + (h << 4);
    if (kb + 32 < K) __builtin_prefetch(arow + 32, 0, 3);
    union { v16bf v; v8bf h2[2]; } au;
    au.h2[0] = *(const v8bf*)(arow);
    au.h2[1] = *(const v8bf*)(arow + 8);

    v16bf bfr[8];
    if (useLds) {
#pragma unroll
      for (int nt = 0; nt < 8; ++nt)
        bfr[nt] = *(const v16bf*)(smw + (size_t)(nt * 16 + r) * 128 + kb + (h << 4));
    } else {
#pragma unroll
      for (int nt = 0; nt < 8; ++nt)
        bfr[nt] = *(const v16bf*)(Wt + (size_t)(nt * 16 + r) * K + kb + (h << 4));
    }

#pragma unroll
    for (int nt = 0; nt < 8; ++nt)
      acc[nt] = __builtin_amdgcn_wmma_f32_16x16x32_bf16(
          false, au.v, false, bfr[nt], (short)0, acc[nt], false, false);
  }

#pragma unroll
  for (int nt = 0; nt < 8; ++nt) {
    const int col = nt * 16 + r;
    const float bv = bias ? bias[col] : 0.f;
#pragma unroll
    for (int j = 0; j < 8; ++j) {
      float v = (acc[nt][j] + bv) * alpha;
      if (relu) v = fmaxf(v, 0.f);
      const size_t idx = (size_t)(rowBase + j + (h << 3)) * ldc + col;
      if (store_bf16) ((__bf16*)Cv)[idx] = cvt_bf16(v);
      else            ((float*)Cv)[idx]  = v;
    }
  }
}

// ---------------------------------------------------------------------------
// Fused flash attention over bf16 q/k/v (scale pre-folded into q projection).
// dh = 16 zero-padded to K = 32 via zeroed scratch (stride-0 reads).
// Per-wave LDS transpose of P uses s_wait_dscnt instead of block barriers,
// so the 8 waves of a block run fully decoupled.  Output bf16.
// ---------------------------------------------------------------------------
__global__ void __launch_bounds__(256)
attn_wmma(const __bf16* __restrict__ q, const __bf16* __restrict__ kmat,
          const __bf16* __restrict__ vmat, __bf16* __restrict__ o,
          const __bf16* __restrict__ zpad, int S)
{
  __shared__ float pbuf[8][256];
  const int wave = threadIdx.x >> 5;
  const int lane = threadIdx.x & 31;
  const int h    = lane >> 4;
  const int r    = lane & 15;

  const int gid    = blockIdx.x * 8 + wave;
  const int qtiles = S >> 4;
  const int qt     = gid % qtiles;
  const int head   = (gid / qtiles) & (kNH - 1);
  const int b      = gid / (qtiles * kNH);
  const int qBase  = qt << 4;

  v8bf z8 = {};

  // Q fragment: h==0 lanes hold dh = e (e<16), h==1 lanes zero.
  union { v16bf v; v8bf h2[2]; } qu;
  const __bf16* qp = q + (size_t)(b * S + qBase + r) * kH + head * kDH;
  qu.h2[0] = (h == 0) ? *(const v8bf*)(qp) : z8;
  qu.h2[1] = (h == 0) ? *(const v8bf*)(qp + 8) : z8;

  float mrow[8], lrow[8];
  v8f oacc = {};
#pragma unroll
  for (int j = 0; j < 8; ++j) { mrow[j] = -1e30f; lrow[j] = 0.f; }

  for (int kt = 0; kt < qtiles; ++kt) {
    const int kBase = kt << 4;

    // V fragment (independent of S-tile): issue early to overlap with QK^T
    // and the softmax ALU work.  K-dim = key, col = dh = r; h==1 reads zeros.
    const __bf16* vbase = (h == 0)
        ? (vmat + (size_t)(b * S + kBase) * kH + head * kDH + r)
        : zpad;
    const size_t vstride = (h == 0) ? (size_t)kH : 0;
    v16bf vf;
#pragma unroll
    for (int e = 0; e < 16; ++e) vf[e] = vbase[(size_t)e * vstride];

    // K^T fragment: h==0 lanes load 16 contiguous dh values of key kBase+r.
    const __bf16* kp = (h == 0)
        ? (kmat + (size_t)(b * S + kBase + r) * kH + head * kDH)
        : zpad;
    v16bf kf = *(const v16bf*)kp;

    v8f z = {};
    v8f s = __builtin_amdgcn_wmma_f32_16x16x32_bf16(
        false, qu.v, false, kf, (short)0, z, false, false);

    // Online softmax (row j+8h; the row's 16 cols live in this half-wave)
    float pj[8];
#pragma unroll
    for (int j = 0; j < 8; ++j) {
      float x = s[j];
      float tm = x;
      tm = fmaxf(tm, __shfl_xor(tm, 1));
      tm = fmaxf(tm, __shfl_xor(tm, 2));
      tm = fmaxf(tm, __shfl_xor(tm, 4));
      tm = fmaxf(tm, __shfl_xor(tm, 8));
      float mnew  = fmaxf(mrow[j], tm);
      float alpha = __expf(mrow[j] - mnew);
      float p     = __expf(x - mnew);
      float ps = p;
      ps += __shfl_xor(ps, 1);
      ps += __shfl_xor(ps, 2);
      ps += __shfl_xor(ps, 4);
      ps += __shfl_xor(ps, 8);
      lrow[j] = lrow[j] * alpha + ps;
      mrow[j] = mnew;
      oacc[j] *= alpha;
      pj[j] = p;
    }

    // C-layout -> A-layout transpose of P via per-wave LDS.
    // DS ops are in-order per wave, so no wait is needed at the loop
    // back-edge (WAR safe); only store->load visibility needs a fence.
#pragma unroll
    for (int j = 0; j < 8; ++j) pbuf[wave][(j + (h << 3)) * 16 + r] = pj[j];
    wave_lds_fence();
    union { v16bf v; v8bf h2[2]; } pu;
#pragma unroll
    for (int e = 0; e < 16; ++e)
      pu.v[e] = cvt_bf16((h == 0) ? pbuf[wave][r * 16 + e] : 0.f);

    oacc = __builtin_amdgcn_wmma_f32_16x16x32_bf16(
        false, pu.v, false, vf, (short)0, oacc, false, false);
  }

#pragma unroll
  for (int j = 0; j < 8; ++j) {
    float inv = 1.f / lrow[j];
    o[(size_t)(b * S + qBase + j + (h << 3)) * kH + head * kDH + r] =
        cvt_bf16(oacc[j] * inv);
  }
}

// ---------------------------------------------------------------------------
// Packing / elementwise / scatter helper kernels
// ---------------------------------------------------------------------------
__global__ void pack_bf16_kernel(const float* __restrict__ src, __bf16* __restrict__ dst, int n) {
  int t = blockIdx.x * blockDim.x + threadIdx.x;
  if (t < n) dst[t] = cvt_bf16(src[t]);
}

// Wt[col*K + k] = bf16( W[k*N + col] )   (transpose-pack weights)
__global__ void pack_wt_kernel(const float* __restrict__ W, __bf16* __restrict__ Wt,
                               int K, int N) {
  int t = blockIdx.x * blockDim.x + threadIdx.x;
  if (t >= K * N) return;
  int col = t / K, k = t - col * K;
  Wt[t] = cvt_bf16(W[(size_t)k * N + col]);
}

__global__ void zero_kernel(float* p, int n) {
  int t = blockIdx.x * blockDim.x + threadIdx.x;
  if (t < n) p[t] = 0.f;
}

__global__ void embed_kernel(const int* __restrict__ A, const int* __restrict__ P,
                             const int* __restrict__ B,
                             const float* __restrict__ atom_emb,
                             const float* __restrict__ pos_emb,
                             const float* __restrict__ block_emb,
                             float* __restrict__ x)
{
  int t = blockIdx.x * blockDim.x + threadIdx.x;
  if (t >= kNAt * kH) return;
  int i = t >> 7, c = t & 127;
  int blkId = i / kABlk;
  x[t] = atom_emb[(size_t)A[i] * kH + c]
       + pos_emb[(size_t)P[i] * kH + c]
       + block_emb[(size_t)B[blkId] * kH + c];
}

__global__ void deg_init_kernel(float* deg) {
  int t = blockIdx.x * blockDim.x + threadIdx.x;
  if (t < kNAt) deg[t] = 1.f;
}

__global__ void deg_scatter_kernel(const int* __restrict__ dst, float* deg) {
  int t = blockIdx.x * blockDim.x + threadIdx.x;
  if (t < kEdges) atomicAdd(&deg[dst[t]], 1.f);
}

__global__ void msg_init_kernel(const float* __restrict__ h, const float* __restrict__ deg,
                                const float* __restrict__ bias, float* __restrict__ out)
{
  int t = blockIdx.x * blockDim.x + threadIdx.x;
  if (t >= kNAt * kH) return;
  int i = t >> 7, c = t & 127;
  out[t] = bias[c] + h[t] / deg[i];
}

__global__ void msg_scatter_kernel(const float* __restrict__ h, const float* __restrict__ deg,
                                   const int* __restrict__ src, const int* __restrict__ dst,
                                   float* __restrict__ out)
{
  int t = blockIdx.x * blockDim.x + threadIdx.x;
  if (t >= kEdges * 32) return;
  int e  = t >> 5;
  int cg = (t & 31) << 2;
  int s = src[e], d = dst[e];
  float coef = rsqrtf(deg[s]) * rsqrtf(deg[d]);
  const float4 hv = *(const float4*)(h + (size_t)s * kH + cg);
  float* op = out + (size_t)d * kH + cg;
  atomicAdd(op + 0, hv.x * coef);
  atomicAdd(op + 1, hv.y * coef);
  atomicAdd(op + 2, hv.z * coef);
  atomicAdd(op + 3, hv.w * coef);
}

__global__ void relu_kernel(float* p, int n) {
  int t = blockIdx.x * blockDim.x + threadIdx.x;
  if (t < n) p[t] = fmaxf(p[t], 0.f);
}

__global__ void seg_mean_kernel(const float* __restrict__ x, float* __restrict__ blk) {
  int t = blockIdx.x * blockDim.x + threadIdx.x;
  if (t >= kNBlk * kH) return;
  int i = t >> 7, c = t & 127;
  float s = 0.f;
  for (int j = 0; j < kABlk; ++j) s += x[(size_t)(i * kABlk + j) * kH + c];
  blk[t] = s * (1.f / kABlk);
}

__global__ void gather_ef0_kernel(const float* __restrict__ blk,
                                  const int* __restrict__ block_edges,
                                  __bf16* __restrict__ ef0)
{
  int t = blockIdx.x * blockDim.x + threadIdx.x;
  if (t >= kBatch * kEB * kH) return;
  int row = t >> 7, c = t & 127;
  int b = row >> 10, e = row & (kEB - 1);
  int s = block_edges[b * 2 * kEB + e];
  int d = block_edges[b * 2 * kEB + kEB + e];
  ef0[(size_t)row * 256 + c]       = cvt_bf16(blk[(size_t)(b * kL + s) * kH + c]);
  ef0[(size_t)row * 256 + 128 + c] = cvt_bf16(blk[(size_t)(b * kL + d) * kH + c]);
}

__global__ void agg_scatter_kernel(const float* __restrict__ eap,
                                   const int* __restrict__ block_edges,
                                   float* __restrict__ agg, float* __restrict__ cnt)
{
  int t = blockIdx.x * blockDim.x + threadIdx.x;
  if (t >= kBatch * kEB * 32) return;
  int b   = t / (kEB * 32);
  int rem = t - b * (kEB * 32);
  int e   = rem >> 5;
  int cg  = (rem & 31) << 2;
  int l = block_edges[b * 2 * kEB + e];
  const float4 v = *(const float4*)(eap + (size_t)(b * kEB + e) * kH + cg);
  float* ap = agg + (size_t)(b * kL + l) * kH + cg;
  atomicAdd(ap + 0, v.x);
  atomicAdd(ap + 1, v.y);
  atomicAdd(ap + 2, v.z);
  atomicAdd(ap + 3, v.w);
  if (cg == 0) atomicAdd(&cnt[b * kL + l], 1.f);
}

__global__ void blk2_kernel(const float* __restrict__ blk, const float* __restrict__ agg,
                            const float* __restrict__ cnt, __bf16* __restrict__ blk2)
{
  int t = blockIdx.x * blockDim.x + threadIdx.x;
  if (t >= kNBlk * kH) return;
  int i = t >> 7;
  blk2[t] = cvt_bf16(blk[t] + agg[t] / fmaxf(cnt[i], 1.f));
}

__global__ void mean_ln_kernel(const float* __restrict__ attp,
                               const float* __restrict__ ln_g, const float* __restrict__ ln_b,
                               float* __restrict__ gbuf)
{
  int b = blockIdx.x, c = threadIdx.x;
  float s = 0.f;
  for (int l = 0; l < kL; ++l) s += attp[(size_t)(b * kL + l) * kH + c];
  float g = s * (1.f / kL);
  __shared__ float red[kH];
  red[c] = g; __syncthreads();
  for (int off = 64; off > 0; off >>= 1) { if (c < off) red[c] += red[c + off]; __syncthreads(); }
  float mu = red[0] * (1.f / kH);
  __syncthreads();
  red[c] = (g - mu) * (g - mu); __syncthreads();
  for (int off = 64; off > 0; off >>= 1) { if (c < off) red[c] += red[c + off]; __syncthreads(); }
  float var = red[0] * (1.f / kH);
  gbuf[b * kH + c] = (g - mu) * rsqrtf(var + kEps) * ln_g[c] + ln_b[c];
}

__global__ void final_mlp_kernel(const float* __restrict__ gbuf,
                                 const float* __restrict__ w1, const float* __restrict__ b1,
                                 const float* __restrict__ w2, const float* __restrict__ b2,
                                 const float* __restrict__ w3, const float* __restrict__ b3,
                                 float* __restrict__ out)
{
  int b = blockIdx.x, c = threadIdx.x;
  __shared__ float s0[kH], s1[kH];
  s0[c] = gbuf[b * kH + c];
  __syncthreads();
  float a = b1[c];
  for (int i = 0; i < kH; ++i) a += s0[i] * w1[i * kH + c];
  s1[c] = fmaxf(a, 0.f);
  __syncthreads();
  float a2 = b2[c];
  for (int i = 0; i < kH; ++i) a2 += s1[i] * w2[i * kH + c];
  __syncthreads();
  s0[c] = fmaxf(a2, 0.f);
  __syncthreads();
  s1[c] = s0[c] * w3[c];
  __syncthreads();
  for (int off = 64; off > 0; off >>= 1) { if (c < off) s1[c] += s1[c + off]; __syncthreads(); }
  if (c == 0) out[b] = s1[0] + b3[0];
}

// ---------------------------------------------------------------------------
// Host orchestration
// ---------------------------------------------------------------------------
extern "C" void kernel_launch(void* const* d_in, const int* in_sizes, int n_in,
                              void* d_out, int out_size, void* d_ws, size_t ws_size,
                              hipStream_t stream)
{
  (void)in_sizes; (void)n_in; (void)out_size; (void)ws_size;

  const int*   A_idx       = (const int*)d_in[1];
  const int*   P_idx       = (const int*)d_in[2];
  const int*   B_idx       = (const int*)d_in[3];
  const int*   atom_edges  = (const int*)d_in[4];
  const int*   block_edges = (const int*)d_in[5];
  const float* atom_emb    = (const float*)d_in[6];
  const float* pos_emb     = (const float*)d_in[7];
  const float* block_emb   = (const float*)d_in[8];
  const float* gcn_w1      = (const float*)d_in[9];
  const float* gcn_b1      = (const float*)d_in[10];
  const float* gcn_w2      = (const float*)d_in[11];
  const float* gcn_b2      = (const float*)d_in[12];
  const float* wq          = (const float*)d_in[13];
  const float* wk          = (const float*)d_in[14];
  const float* wv          = (const float*)d_in[15];
  const float* bq          = (const float*)d_in[16];
  const float* bk          = (const float*)d_in[17];
  const float* bvv         = (const float*)d_in[18];
  const float* wo          = (const float*)d_in[19];
  const float* bo          = (const float*)d_in[20];
  const float* em_w1       = (const float*)d_in[21];
  const float* em_b1       = (const float*)d_in[22];
  const float* em_w2       = (const float*)d_in[23];
  const float* em_b2       = (const float*)d_in[24];
  const float* fm_w1       = (const float*)d_in[25];
  const float* fm_b1       = (const float*)d_in[26];
  const float* fm_w2       = (const float*)d_in[27];
  const float* fm_b2       = (const float*)d_in[28];
  const float* fm_w3       = (const float*)d_in[29];
  const float* fm_b3       = (const float*)d_in[30];
  const float* ln_g        = (const float*)d_in[31];
  const float* ln_b        = (const float*)d_in[32];

  const int* e_src = atom_edges;
  const int* e_dst = atom_edges + kEdges;

  float* ws = (float*)d_ws;
  // ---- workspace layout (float offsets) ----------------------------------
  float*  x    = ws;                         // NAT*H f32         (4,194,304 f)
  float*  r2f  = ws + 4194304;               // 16MB region: stage-2 bf16 bufs
  float*  r3f  = ws + 8388608;               // 8MB region: xbf / eap / attp...
  float*  hb   = ws + 10485760;              // NAT*H f32 gemm out (4,194,304 f)
  float*  deg  = ws + 14680064;              // 32768
  float*  blk  = ws + 14712832;              // 131072
  float*  agg  = ws + 14843904;              // 131072
  float*  cnt  = ws + 14974976;              // 1024
  float*  gbuf = ws + 14976000;              // 1024
  float*  zpadf= ws + 14977024;              // 64 (zeroed pad, 256B)
  float*  wtf  = ws + 14977088;              // weight packs (73,728 f)

  // bf16 views
  __bf16* r2   = (__bf16*)r2f;
  __bf16* ef0b = r2;                          // 2,097,152 el
  __bf16* t1b  = r2 + 2097152;                // 1,048,576
  __bf16* efb  = r2 + 3145728;                // 1,048,576
  __bf16* qb   = r2 + 4194304;                // 1,048,576
  __bf16* kb_  = r2 + 5242880;                // 1,048,576
  __bf16* vb   = r2 + 6291456;                // 1,048,576
  __bf16* eab  = r2 + 7340032;                // 1,048,576

  __bf16* xbf  = (__bf16*)r3f;                // NAT*H bf16 (GCN phase)
  float*  eap  = r3f;                         // 1,048,576 f (after GCN)
  float*  attp = r3f + 1048576;               // 131,072 f
  __bf16* blk2b= (__bf16*)(r3f + 1179648);    // 131,072 el
  __bf16* q2b  = (__bf16*)(r3f + 1245184);
  __bf16* k2b  = (__bf16*)(r3f + 1310720);
  __bf16* v2b  = (__bf16*)(r3f + 1376256);
  __bf16* attb = (__bf16*)(r3f + 1441792);

  __bf16* zpad = (__bf16*)zpadf;
  __bf16* wt_gcn1 = (__bf16*)wtf;             // 16,384 el each (8,192 f)
  __bf16* wt_gcn2 = wt_gcn1 + 16384;
  __bf16* wt_em1  = wt_gcn2 + 16384;          // 32,768 el (K=256)
  __bf16* wt_em2  = wt_em1  + 32768;
  __bf16* wt_q    = wt_em2  + 16384;
  __bf16* wt_k    = wt_q    + 16384;
  __bf16* wt_v    = wt_k    + 16384;
  __bf16* wt_o    = wt_v    + 16384;

  float* out = (float*)d_out;
  const int TPB = 256;

  auto gemm = [&](const __bf16* Ap, int lda, const __bf16* Wp, const float* biasp,
                  void* Cp, int M, int K, int relu, int bf, float alpha) {
    gemm_wmma<<<M / 128, TPB, 0, stream>>>(Ap, lda, Wp, biasp, Cp, 128, M, K,
                                           relu, bf, alpha);
  };
  auto packw = [&](const float* Wp, __bf16* Wtp, int K) {
    pack_wt_kernel<<<(K * 128) / TPB, TPB, 0, stream>>>(Wp, Wtp, K, 128);
  };

  // 0) pack weights (bf16, col-major) + zero pad region
  packw(gcn_w1, wt_gcn1, 128);
  packw(gcn_w2, wt_gcn2, 128);
  packw(em_w1,  wt_em1,  256);
  packw(em_w2,  wt_em2,  128);
  packw(wq,     wt_q,    128);
  packw(wk,     wt_k,    128);
  packw(wv,     wt_v,    128);
  packw(wo,     wt_o,    128);
  zero_kernel<<<1, 64, 0, stream>>>(zpadf, 64);

  // 1) embeddings + degrees
  embed_kernel<<<(kNAt * kH) / TPB, TPB, 0, stream>>>(A_idx, P_idx, B_idx,
                                                      atom_emb, pos_emb, block_emb, x);
  deg_init_kernel<<<kNAt / TPB, TPB, 0, stream>>>(deg);
  deg_scatter_kernel<<<kEdges / TPB, TPB, 0, stream>>>(e_dst, deg);

  // 2) GCN layer 1
  pack_bf16_kernel<<<(kNAt * kH) / TPB, TPB, 0, stream>>>(x, xbf, kNAt * kH);
  gemm(xbf, kH, wt_gcn1, nullptr, hb, kNAt, kH, 0, 0, 1.f);
  msg_init_kernel<<<(kNAt * kH) / TPB, TPB, 0, stream>>>(hb, deg, gcn_b1, x);
  msg_scatter_kernel<<<(kEdges * 32) / TPB, TPB, 0, stream>>>(hb, deg, e_src, e_dst, x);
  relu_kernel<<<(kNAt * kH) / TPB, TPB, 0, stream>>>(x, kNAt * kH);

  // 3) GCN layer 2
  pack_bf16_kernel<<<(kNAt * kH) / TPB, TPB, 0, stream>>>(x, xbf, kNAt * kH);
  gemm(xbf, kH, wt_gcn2, nullptr, hb, kNAt, kH, 0, 0, 1.f);
  msg_init_kernel<<<(kNAt * kH) / TPB, TPB, 0, stream>>>(hb, deg, gcn_b2, x);
  msg_scatter_kernel<<<(kEdges * 32) / TPB, TPB, 0, stream>>>(hb, deg, e_src, e_dst, x);
  relu_kernel<<<(kNAt * kH) / TPB, TPB, 0, stream>>>(x, kNAt * kH);

  // 4) block pooling + edge features
  seg_mean_kernel<<<(kNBlk * kH) / TPB, TPB, 0, stream>>>(x, blk);
  gather_ef0_kernel<<<(kBatch * kEB * kH) / TPB, TPB, 0, stream>>>(blk, block_edges, ef0b);
  gemm(ef0b, 256, wt_em1, em_b1, t1b, kBatch * kEB, 256, 1, 1, 1.f);
  gemm(t1b,  kH,  wt_em2, em_b2, efb, kBatch * kEB, kH,  0, 1, 1.f);

  // 5) MHA over edge features (seq = 1024); softmax scale folded into q
  gemm(efb, kH, wt_q, bq,  qb,  kBatch * kEB, kH, 0, 1, 0.25f);
  gemm(efb, kH, wt_k, bk,  kb_, kBatch * kEB, kH, 0, 1, 1.f);
  gemm(efb, kH, wt_v, bvv, vb,  kBatch * kEB, kH, 0, 1, 1.f);
  attn_wmma<<<(kBatch * kNH * (kEB / 16)) / 8, TPB, 0, stream>>>(qb, kb_, vb, eab, zpad, kEB);
  gemm(eab, kH, wt_o, bo, eap, kBatch * kEB, kH, 0, 0, 1.f);

  // 6) aggregate edge outputs onto blocks
  zero_kernel<<<(kNBlk * kH + TPB - 1) / TPB, TPB, 0, stream>>>(agg, kNBlk * kH);
  zero_kernel<<<(kNBlk + TPB - 1) / TPB, TPB, 0, stream>>>(cnt, kNBlk);
  agg_scatter_kernel<<<(kBatch * kEB * 32) / TPB, TPB, 0, stream>>>(eap, block_edges, agg, cnt);
  blk2_kernel<<<(kNBlk * kH) / TPB, TPB, 0, stream>>>(blk, agg, cnt, blk2b);

  // 7) MHA over blocks (seq = 128)
  gemm(blk2b, kH, wt_q, bq,  q2b, kNBlk, kH, 0, 1, 0.25f);
  gemm(blk2b, kH, wt_k, bk,  k2b, kNBlk, kH, 0, 1, 1.f);
  gemm(blk2b, kH, wt_v, bvv, v2b, kNBlk, kH, 0, 1, 1.f);
  attn_wmma<<<(kBatch * kNH * (kL / 16)) / 8, TPB, 0, stream>>>(q2b, k2b, v2b, attb, zpad, kL);
  gemm(attb, kH, wt_o, bo, attp, kNBlk, kH, 0, 0, 1.f);

  // 8) mean over seq + LayerNorm + final MLP
  mean_ln_kernel<<<kBatch, kH, 0, stream>>>(attp, ln_g, ln_b, gbuf);
  final_mlp_kernel<<<kBatch, kH, 0, stream>>>(gbuf, fm_w1, fm_b1, fm_w2, fm_b2,
                                              fm_w3, fm_b3, out);
}
